// SimpleNTXentLoss_15676630630501
// MI455X (gfx1250) — matrix operand order repaired and checked by reading, not compiled, not run on hardware
//
#include <hip/hip_runtime.h>
#include <hip/hip_bf16.h>

typedef __attribute__((ext_vector_type(4)))  _Float16 v4h;
typedef __attribute__((ext_vector_type(8)))  _Float16 v8h;
typedef __attribute__((ext_vector_type(16))) _Float16 v16h;
typedef __attribute__((ext_vector_type(8)))  float    v8f;

#define B_N 8192
#define D_N 128
#define C_N 16
#define TEMP_INV 2.0f   // 1 / TEMP, TEMP = 0.5

static __device__ __forceinline__ v16h cat8(v8h lo, v8h hi) {
    return __builtin_shufflevector(lo, hi, 0,1,2,3,4,5,6,7,8,9,10,11,12,13,14,15);
}

// Async global->LDS copy of 16 bytes (plus a second 16B at +4096 on both sides).
// INST_OFFSET is added to BOTH the LDS destination and the global source
// (ISA 08_async_tensor 4.4), tracked with ASYNCcnt.
static __device__ __forceinline__ void async_copy_b128(uint32_t lds_addr, uint64_t gaddr) {
    asm volatile("global_load_async_to_lds_b128 %0, %1, off"
                 :: "v"(lds_addr), "v"(gaddr) : "memory");
    asm volatile("global_load_async_to_lds_b128 %0, %1, off offset:4096"
                 :: "v"(lds_addr), "v"(gaddr) : "memory");
}

static __device__ __forceinline__ void wait_async_le2() {
    asm volatile("s_wait_asynccnt 0x2" ::: "memory");
}

// ---------------- prep kernels ----------------

__global__ void zero_kernel(float* p) {
    p[threadIdx.x] = 0.0f;   // 32 threads: gLsum[16] + gVcnt[16]
}

__global__ void conv_emb_kernel(const float* __restrict__ emb, _Float16* __restrict__ embH) {
    int t = blockIdx.x * blockDim.x + threadIdx.x;   // 262144 threads, 4 elems each
    float4 f = ((const float4*)emb)[t];
    v4h h = {(_Float16)f.x, (_Float16)f.y, (_Float16)f.z, (_Float16)f.w};
    ((v4h*)embH)[t] = h;
}

__global__ void conv_lab_kernel(const int* __restrict__ labels, _Float16* __restrict__ labTH) {
    int t = blockIdx.x * blockDim.x + threadIdx.x;   // 131072 threads
    int j = t & (B_N - 1);
    int c = t >> 13;
    labTH[(size_t)c * B_N + j] = (_Float16)labels[(size_t)j * C_N + c];
}

__global__ void cnt1_kernel(const int* __restrict__ labels, float* __restrict__ cnt1) {
    __shared__ float red[256];
    const int c = blockIdx.x;       // one block per class
    const int t = threadIdx.x;
    float s = 0.0f;
    for (int j = t; j < B_N; j += 256) s += (float)labels[(size_t)j * C_N + c];
    red[t] = s;
    __syncthreads();
    for (int off = 128; off > 0; off >>= 1) {
        if (t < off) red[t] += red[t + off];
        __syncthreads();
    }
    if (t == 0) cnt1[c] = red[0];
}

// ---------------- main fused kernel ----------------
// One wave owns one 16-row i-tile (embI resident in VGPRs as WMMA B operands).
// The workgroup cooperatively double-buffers 32x128 f16 embJ slabs in LDS via
// async global->LDS loads (ASYNCcnt); WMMA A fragments come from ds_load_b128.
// sim^T tile -> exp (diag zeroed) -> f16 pack lands natively in the 16-bit
// A-matrix layout of a second WMMA chain accumulating S1[i][c].

__global__ __launch_bounds__(256)
void ntxent_main_kernel(const _Float16* __restrict__ embH,
                        const _Float16* __restrict__ labTH,
                        const int*      __restrict__ labels,
                        const float*    __restrict__ cnt1,
                        float* __restrict__ gLsum,
                        float* __restrict__ gVcnt)
{
    __shared__ _Float16 shbuf[2][32 * D_N];   // 2 x 8 KB double buffer

    const int tid   = threadIdx.x;
    const int lane  = tid & 31;
    const int wave  = tid >> 5;
    const int hl    = lane >> 4;     // half-of-wave: 0 (lanes 0-15) or 1 (16-31)
    const int ln    = lane & 15;
    const int ibase = (blockIdx.x * 8 + wave) * 16;
    const int i_n   = ibase + ln;    // N-index (column of sim^T) for this lane

    const char* embB = (const char*)embH;

    // embI as four K=32 B-matrices (B[K=k][N=i] = emb[ibase+i][k]), resident in VGPRs.
    v16h Bm[4];
    {
        const _Float16* rowp = embH + (size_t)i_n * D_N + hl * 8;
        #pragma unroll
        for (int kb = 0; kb < 4; ++kb) {
            const v8h* p = (const v8h*)(rowp + kb * 32);
            Bm[kb] = cat8(p[0], p[2]);
        }
    }

    // Prologue: stage slab 0 into buffer 0 (2 async b128 per thread = 8 KB/block).
    async_copy_b128((uint32_t)(uintptr_t)&shbuf[0][0] + tid * 16,
                    (uint64_t)(uintptr_t)(embB + tid * 16));

    v8f   S1    = {};     // accumulated D[M = i - ibase][N = c] across all j
    float denom = 0.0f;   // this lane's partial row-sum for i = i_n

    const _Float16* labRow = labTH + (size_t)ln * B_N;

    for (int jt = 0; jt < B_N; jt += 32) {
        const int cbuf = (jt >> 5) & 1;
        const int nbuf = cbuf ^ 1;
        const int jnext = (jt + 32) & (B_N - 1);   // wrap: last prefetch hits dead buffer

        // Prefetch next slab while computing on the current one.
        async_copy_b128((uint32_t)(uintptr_t)&shbuf[nbuf][0] + tid * 16,
                        (uint64_t)(uintptr_t)(embB + (size_t)jnext * 256 + tid * 16));
        // Async loads complete in order: <=2 outstanding means current slab done.
        wait_async_le2();
        __syncthreads();

        const _Float16* slab = &shbuf[cbuf][0];
        v8f d0 = {};
        v8f d1 = {};
        #pragma unroll
        for (int kb = 0; kb < 4; ++kb) {
            const _Float16* a0p = slab + ln * D_N + kb * 32 + hl * 8;         // tile0 row
            const _Float16* a1p = slab + (16 + ln) * D_N + kb * 32 + hl * 8;  // tile1 row
            v16h A0 = cat8(*(const v8h*)a0p, *(const v8h*)(a0p + 16));
            v16h A1 = cat8(*(const v8h*)a1p, *(const v8h*)(a1p + 16));
            d0 = __builtin_amdgcn_wmma_f32_16x16x32_f16(false, A0, false, Bm[kb],
                                                        (short)0, d0, false, false);
            d1 = __builtin_amdgcn_wmma_f32_16x16x32_f16(false, A1, false, Bm[kb],
                                                        (short)0, d1, false, false);
        }

        // exp(sim/TEMP), zero the diagonal, accumulate denom, pack f16.
        // d0[v] is (j = jt + hl*8 + v, i = i_n); d1[v] is j+16. The packed E is
        // exactly the 16-bit A-matrix layout (M=i, K=j-jt) for the S1 WMMA.
        v16h E;
        const int jb0 = jt + hl * 8;
        #pragma unroll
        for (int v = 0; v < 8; ++v) {
            const int j0 = jb0 + v;
            const int j1 = j0 + 16;
            float e0 = __expf(d0[v] * TEMP_INV);
            float e1 = __expf(d1[v] * TEMP_INV);
            e0 = (j0 == i_n) ? 0.0f : e0;
            e1 = (j1 == i_n) ? 0.0f : e1;
            denom += e0 + e1;
            E[v]     = (_Float16)e0;
            E[v + 8] = (_Float16)e1;
        }

        // Label B-matrix: B[K=j-jt][N=c] = labTH[c][jt + k]; lane holds c = ln.
        const _Float16* lp = labRow + jt + hl * 8;
        v16h LB = cat8(*(const v8h*)lp, *(const v8h*)(lp + 16));

        S1 = __builtin_amdgcn_wmma_f32_16x16x32_f16(false, E, false, LB,
                                                    (short)0, S1, false, false);

        // All waves must finish reading this buffer before it is refilled.
        __syncthreads();
    }

    // denom: lanes l and l+16 cover disjoint j halves for the same i = ln.
    float denom_full = denom + __shfl_xor(denom, 16, 32);

    // Epilogue: lane holds S1[ibase + v + 8*hl][c = ln] for v = 0..7.
    float lsum = 0.0f, vsum = 0.0f;
    const float c1 = cnt1[ln];
    #pragma unroll
    for (int v = 0; v < 8; ++v) {
        const int irow = ibase + v + 8 * hl;
        const float dd = __shfl(denom_full, v + 8 * hl, 32);  // denom for row irow
        const int   lab = labels[(size_t)irow * C_N + ln];
        const float s   = S1[v];
        float pos = lab ? s : (dd - s);
        pos = fmaxf(pos, 1e-8f);
        const float lossv = -__logf(pos / (dd + 1e-8f));
        const float same  = lab ? (c1 - 1.0f) : ((float)B_N - c1 - 1.0f);
        const float vm    = (same > 0.0f) ? 1.0f : 0.0f;
        lsum += lossv * vm;
        vsum += vm;
    }
    lsum += __shfl_xor(lsum, 16, 32);
    vsum += __shfl_xor(vsum, 16, 32);
    if (lane < 16) {
        atomicAdd(&gLsum[ln], lsum);
        atomicAdd(&gVcnt[ln], vsum);
    }
}

// ---------------- finalize ----------------

__global__ void final_kernel(const float* __restrict__ cnt1,
                             const float* __restrict__ gLsum,
                             const float* __restrict__ gVcnt,
                             float* __restrict__ out)
{
    const int c = threadIdx.x;
    float per = 0.0f;
    if (c < C_N) {
        const float c1 = cnt1[c];
        per = gLsum[c] / fmaxf(gVcnt[c], 1.0f);
        if (c1 == 0.0f || c1 == (float)B_N) per = 0.0f;
    }
    for (int off = 8; off > 0; off >>= 1) per += __shfl_down(per, off, 32);
    if (c == 0) out[0] = per / (float)C_N;
}

// ---------------- launcher ----------------

extern "C" void kernel_launch(void* const* d_in, const int* in_sizes, int n_in,
                              void* d_out, int out_size, void* d_ws, size_t ws_size,
                              hipStream_t stream) {
    const float* emb    = (const float*)d_in[0];   // [8192,128] f32
    const int*   labels = (const int*)d_in[1];     // [8192,16]  i32
    float*       out    = (float*)d_out;           // scalar

    char* ws = (char*)d_ws;
    _Float16* embH  = (_Float16*)ws;                              // 2 MB
    _Float16* labTH = (_Float16*)(ws + (size_t)2 * 1024 * 1024);  // 256 KB
    float*    cnt1  = (float*)(ws + (size_t)2 * 1024 * 1024 + 256 * 1024);
    float*    gLsum = cnt1 + 16;
    float*    gVcnt = cnt1 + 32;

    zero_kernel<<<1, 32, 0, stream>>>(gLsum);                 // zeros gLsum+gVcnt
    conv_emb_kernel<<<1024, 256, 0, stream>>>(emb, embH);     // 8192*128/4 threads
    conv_lab_kernel<<<512, 256, 0, stream>>>(labels, labTH);  // 16*8192 threads
    cnt1_kernel<<<16, 256, 0, stream>>>(labels, cnt1);
    ntxent_main_kernel<<<64, 256, 0, stream>>>(embH, labTH, labels, cnt1, gLsum, gVcnt);
    final_kernel<<<1, 32, 0, stream>>>(cnt1, gLsum, gVcnt, out);
}
